// GCNDiehlq1_22763326669148
// MI455X (gfx1250) — matrix operand order repaired and checked by reading, not compiled, not run on hardware
//
#include <hip/hip_runtime.h>
#include <hip/hip_bf16.h>

// Problem constants (match reference)
#define NN 20480
#define EE 327680
#define E2 524288      // next pow2 >= EE for bitonic sort
#define GG 64
#define HF 128
#define CC 6

typedef __attribute__((ext_vector_type(16))) _Float16 v16h;
typedef __attribute__((ext_vector_type(8)))  float    v8f;

static inline int cdiv(int a, int b) { return (a + b - 1) / b; }

// ---- float <-> order-preserving uint -------------------------------------
__device__ inline unsigned ordf(float f) {
  unsigned u = __float_as_uint(f);
  return (u & 0x80000000u) ? ~u : (u | 0x80000000u);
}
__device__ inline float unordf(unsigned u) {
  unsigned b = (u & 0x80000000u) ? (u & 0x7fffffffu) : ~u;
  return __uint_as_float(b);
}

// ---- fills / copies -------------------------------------------------------
__global__ void k_fill_f32(float* p, float v, int n) {
  int t = blockIdx.x * blockDim.x + threadIdx.x; if (t < n) p[t] = v;
}
__global__ void k_fill_u32(unsigned* p, unsigned v, int n) {
  int t = blockIdx.x * blockDim.x + threadIdx.x; if (t < n) p[t] = v;
}
__global__ void k_fill_i32(int* p, int v, int n) {
  int t = blockIdx.x * blockDim.x + threadIdx.x; if (t < n) p[t] = v;
}
__global__ void k_fill_u8(unsigned char* p, unsigned char v, int n) {
  int t = blockIdx.x * blockDim.x + threadIdx.x; if (t < n) p[t] = v;
}
__global__ void k_copy_f32(float* d, const float* s, int n) {
  int t = blockIdx.x * blockDim.x + threadIdx.x; if (t < n) d[t] = s[t];
}
__global__ void k_copy_i32(int* d, const int* s, int n) {
  int t = blockIdx.x * blockDim.x + threadIdx.x; if (t < n) d[t] = s[t];
}

// ---- weight pack: W[Nout=128, K=128 slice] f32 -> WMMA B-fragment f16 -----
// frag[((kt*8+nt)*32+lane)*16 + j] = W[(nt*16 + lane&15)*rowStride + colOff + kt*32 + (lane>>4)*8 + (j&7) + (j>>3)*16]
__global__ void k_pack(const float* __restrict__ W, int rowStride, int colOff,
                       _Float16* __restrict__ out) {
  int t = blockIdx.x * blockDim.x + threadIdx.x;
  if (t >= 16384) return;
  int j    = t & 15;
  int lane = (t >> 4) & 31;
  int nt   = (t >> 9) & 7;
  int kt   = t >> 12;
  int n = nt * 16 + (lane & 15);
  int k = kt * 32 + ((lane >> 4) << 3) + (j & 7) + ((j >> 3) << 4);
  out[t] = (_Float16)W[n * rowStride + colOff + k];
}

// ---- WMMA GEMM: OUT[M,128] = A1@W1^T (+ A2@W2^T) + bias, optional relu ----
// K = 128 per input; weights pre-packed by k_pack. One wave per 16x16 tile,
// 8 waves per block cover all 8 column tiles of one 16-row stripe.
__global__ __launch_bounds__(256) void k_gemm128(
    const float* __restrict__ A1, const _Float16* __restrict__ W1,
    const float* __restrict__ A2, const _Float16* __restrict__ W2,
    const float* __restrict__ bias, float* __restrict__ OUT,
    int M, int relu) {
  const int lane  = threadIdx.x & 31;
  const int ntile = threadIdx.x >> 5;   // 0..7
  const int mtile = blockIdx.x;
  const int m     = mtile * 16 + (lane & 15);
  const int kgrp  = lane >> 4;          // 0 or 1
  v8f acc = {};
  for (int pass = 0; pass < 2; ++pass) {
    const float*     A  = pass ? A2 : A1;
    const _Float16*  Wf = pass ? W2 : W1;
    if (A == nullptr) continue;         // uniform branch
#pragma unroll
    for (int kt = 0; kt < 4; ++kt) {
      v16h a;
#pragma unroll
      for (int q = 0; q < 16; ++q) a[q] = (_Float16)0.0f;
      if (m < M) {
        // 16-bit A layout: halves 0..7 -> K = kb..kb+7 ; halves 8..15 -> kb+16..kb+23
        const float* ap = A + (size_t)m * HF + kt * 32 + kgrp * 8;
        float4 f0 = *(const float4*)(ap);
        float4 f1 = *(const float4*)(ap + 4);
        float4 f2 = *(const float4*)(ap + 16);
        float4 f3 = *(const float4*)(ap + 20);
        a[0]=(_Float16)f0.x; a[1]=(_Float16)f0.y; a[2]=(_Float16)f0.z; a[3]=(_Float16)f0.w;
        a[4]=(_Float16)f1.x; a[5]=(_Float16)f1.y; a[6]=(_Float16)f1.z; a[7]=(_Float16)f1.w;
        a[8]=(_Float16)f2.x; a[9]=(_Float16)f2.y; a[10]=(_Float16)f2.z; a[11]=(_Float16)f2.w;
        a[12]=(_Float16)f3.x; a[13]=(_Float16)f3.y; a[14]=(_Float16)f3.z; a[15]=(_Float16)f3.w;
        __builtin_prefetch(ap + HF, 0, 1);  // global_prefetch_b8 of next row
      }
      v16h b = *(const v16h*)(Wf + ((size_t)((kt * 8 + ntile) * 32 + lane) << 4));
      acc = __builtin_amdgcn_wmma_f32_16x16x32_f16(
          false, a, false, b, (short)0, acc, false, false);
    }
  }
  const int col = ntile * 16 + (lane & 15);
  const float bv = bias ? bias[col] : 0.0f;
#pragma unroll
  for (int r = 0; r < 8; ++r) {
    int row = mtile * 16 + r + kgrp * 8;  // C/D layout: VGPR r -> M=r / M=8+r
    if (row < M) {
      float v = acc[r] + bv;
      if (relu) v = fmaxf(v, 0.0f);
      OUT[(size_t)row * HF + col] = v;
    }
  }
}

// ---- SAGE message scatter: S[dst] += x[src], cnt[dst]++ (sink row = NN) ---
__global__ void k_edge_scatter(const int* __restrict__ src, const int* __restrict__ dst,
                               const float* __restrict__ x, float* __restrict__ S,
                               float* __restrict__ cnt) {
  int e = blockIdx.x;        // one block (128 threads) per edge
  int f = threadIdx.x;
  int s = src[e], d = dst[e];
  atomicAdd(&S[(size_t)d * HF + f], x[(size_t)s * HF + f]);
  if (f == 0) atomicAdd(&cnt[d], 1.0f);
}
__global__ void k_mean(float* S, const float* cnt) {
  int t = blockIdx.x * blockDim.x + threadIdx.x;
  if (t >= NN * HF) return;
  S[t] /= fmaxf(cnt[t >> 7], 1.0f);
}

// ---- edge score: raw[e] = Wp[0:128].h[src] + Wp[128:256].h[dsafe] + bp ----
__global__ __launch_bounds__(256) void k_edge_score(
    const float* __restrict__ h, const int* __restrict__ src,
    const int* __restrict__ dst, const float* __restrict__ Wp,
    const float* __restrict__ bp, float* __restrict__ raw) {
  int lane = threadIdx.x & 31;
  int e = blockIdx.x * 8 + (threadIdx.x >> 5);   // 8 edges per block, uniform per wave
  int s = src[e];
  int dv = dst[e];
  int d = dv < NN ? dv : NN - 1;
  float4 a  = ((const float4*)(h + (size_t)s * HF))[lane];
  float4 b  = ((const float4*)(h + (size_t)d * HF))[lane];
  float4 wa = ((const float4*)Wp)[lane];
  float4 wb = ((const float4*)(Wp + HF))[lane];
  float p = a.x*wa.x + a.y*wa.y + a.z*wa.z + a.w*wa.w
          + b.x*wb.x + b.y*wb.y + b.z*wb.z + b.w*wb.w;
  for (int off = 16; off; off >>= 1) p += __shfl_down(p, off, 32);
  if (lane == 0) raw[e] = p + bp[0];
}

// ---- segment softmax over dst ---------------------------------------------
__global__ void k_seg_max(const float* raw, const int* dst, unsigned* mu) {
  int e = blockIdx.x * blockDim.x + threadIdx.x; if (e >= EE) return;
  int d = dst[e]; int seg = d < NN ? d : NN;
  atomicMax(&mu[seg], ordf(raw[e]));
}
__global__ void k_seg_exp(const float* raw, const int* dst, const unsigned* mu,
                          float* z, float* esc) {
  int e = blockIdx.x * blockDim.x + threadIdx.x; if (e >= EE) return;
  int d = dst[e]; int seg = d < NN ? d : NN;
  float ex = expf(raw[e] - unordf(mu[seg]));
  esc[e] = ex;
  atomicAdd(&z[seg], ex);
}
__global__ void k_seg_norm(const float* esc, const int* dst, const float* z,
                           float* score) {
  int e = blockIdx.x * blockDim.x + threadIdx.x; if (e >= EE) return;
  int d = dst[e]; int seg = d < NN ? d : NN;
  score[e] = esc[e] / z[seg] + 0.5f;
}

// ---- sorting --------------------------------------------------------------
__global__ void k_build_skey(const float* score, const int* dst,
                             unsigned* key, int* idx) {
  int j = blockIdx.x * blockDim.x + threadIdx.x; if (j >= E2) return;
  if (j < EE) {
    bool v = dst[j] < NN;
    float kf = v ? -score[j] : __builtin_inff();
    key[j] = ordf(kf);
    idx[j] = j;
  } else { key[j] = 0xffffffffu; idx[j] = -1; }
}
__global__ void k_bitonic(unsigned* key, int* idx, int jj, int kk) {
  int i = blockIdx.x * blockDim.x + threadIdx.x;
  int ixj = i ^ jj;
  if (ixj > i) {
    unsigned ki = key[i], kx = key[ixj];
    bool up = ((i & kk) == 0);
    if (up ? (ki > kx) : (ki < kx)) {
      key[i] = kx; key[ixj] = ki;
      int t = idx[i]; idx[i] = idx[ixj]; idx[ixj] = t;
    }
  }
}

// ---- sequential greedy matching (matches reference lax.scan) --------------
__global__ void k_greedy(const int* __restrict__ sidx, const int* __restrict__ src,
                         const int* __restrict__ dst, const float* __restrict__ score,
                         unsigned char* __restrict__ rem, int* __restrict__ cluster,
                         float* __restrict__ cscore) {
  if (threadIdx.x || blockIdx.x) return;
  int cnt = 0;
  for (int j = 0; j < EE; ++j) {
    int e = sidx[j];
    int dv = dst[e];
    if (dv >= NN) continue;            // invalid edge
    int s = src[e];
    if (rem[s] && rem[dv]) {
      cluster[s] = cnt; cluster[dv] = cnt;
      rem[s] = 0; rem[dv] = 0;
      cscore[cnt] = score[e];
      ++cnt;
    }
  }
  for (int i = 0; i < NN; ++i)
    if (rem[i]) cluster[i] = cnt++;
}

// ---- pooled features / batch ---------------------------------------------
__global__ void k_cluster_scatter(const float* __restrict__ h, const int* __restrict__ cl,
                                  float* __restrict__ nx) {
  int t = blockIdx.x * blockDim.x + threadIdx.x; if (t >= NN * HF) return;
  atomicAdd(&nx[(size_t)cl[t >> 7] * HF + (t & 127)], h[t]);
}
__global__ void k_scale(float* nx, const float* cs) {
  int t = blockIdx.x * blockDim.x + threadIdx.x; if (t >= NN * HF) return;
  nx[t] *= cs[t >> 7];
}
__global__ void k_batch_set(const int* cl, const int* bcur, int* bout) {
  int i = blockIdx.x * blockDim.x + threadIdx.x; if (i >= NN) return;
  bout[cl[i]] = bcur[i];
}

// ---- edge coalesce --------------------------------------------------------
__global__ void k_rekey(const int* src, const int* dst, const int* cl,
                        unsigned* key, int* idx) {
  int j = blockIdx.x * blockDim.x + threadIdx.x; if (j >= E2) return;
  if (j < EE) {
    int dv = dst[j];
    bool v = dv < NN;
    int ds = dv < NN ? dv : NN - 1;
    unsigned cs = v ? (unsigned)cl[src[j]] : 0u;
    unsigned cd = v ? (unsigned)cl[ds] : (unsigned)NN;
    key[j] = cs * (unsigned)(NN + 1) + cd;
    idx[j] = j;
  } else { key[j] = 0xffffffffu; idx[j] = -1; }
}
__global__ void k_dedup(const unsigned* key, int* nsrc, int* ndst) {
  int j = blockIdx.x * blockDim.x + threadIdx.x; if (j >= EE) return;
  unsigned k = key[j];
  bool dup = (j > 0) && (key[j - 1] == k);
  nsrc[j] = dup ? 0  : (int)(k / (NN + 1));
  ndst[j] = dup ? NN : (int)(k % (NN + 1));
}

// ---- graph readout --------------------------------------------------------
__global__ void k_gaccum(const float* h, const int* b, float* gsum, float* gcnt) {
  int t = blockIdx.x * blockDim.x + threadIdx.x; if (t >= NN * HF) return;
  int bb = b[t >> 7];
  if (bb < GG) {
    atomicAdd(&gsum[(size_t)bb * HF + (t & 127)], h[t]);
    if ((t & 127) == 0) atomicAdd(&gcnt[bb], 1.0f);
  }
}
__global__ void k_gmean(const float* gsum, const float* gcnt, float* gv) {
  int t = blockIdx.x * blockDim.x + threadIdx.x; if (t >= GG * HF) return;
  gv[t] = gsum[t] / fmaxf(gcnt[t >> 7], 1.0f);
}

// ---- fc2 + log_softmax ----------------------------------------------------
__global__ void k_fc2(const float* gh, const float* W2, const float* b2, float* out) {
  int g = blockIdx.x, lane = threadIdx.x;
  __shared__ float o[8];
  for (int c = 0; c < CC; ++c) {
    float p = 0.0f;
    for (int f = lane; f < HF; f += 32) p += gh[(size_t)g * HF + f] * W2[(size_t)c * HF + f];
    for (int off = 16; off; off >>= 1) p += __shfl_down(p, off, 32);
    if (lane == 0) o[c] = p + b2[c];
  }
  __syncthreads();
  if (lane == 0) {
    float mx = o[0];
    for (int c = 1; c < CC; ++c) mx = fmaxf(mx, o[c]);
    float se = 0.0f;
    for (int c = 0; c < CC; ++c) se += expf(o[c] - mx);
    float ls = mx + logf(se);
    for (int c = 0; c < CC; ++c) out[g * CC + c] = o[c] - ls;
  }
}

// ===========================================================================
extern "C" void kernel_launch(void* const* d_in, const int* in_sizes, int n_in,
                              void* d_out, int out_size, void* d_ws, size_t ws_size,
                              hipStream_t stream) {
  (void)in_sizes; (void)n_in; (void)out_size; (void)ws_size;
  const float* Xin  = (const float*)d_in[0];
  const int*   EI   = (const int*)d_in[1];   // [2,E]: first E = src, next E = dst
  const int*   Bin  = (const int*)d_in[2];
  const float *Wl[6], *bl[6], *Wr[6];
  for (int j = 0; j < 6; ++j) {
    Wl[j] = (const float*)d_in[3 + 3 * j];
    bl[j] = (const float*)d_in[4 + 3 * j];
    Wr[j] = (const float*)d_in[5 + 3 * j];
  }
  const float *Wlin[3], *blin[3], *Wpp[3], *bpp[3];
  for (int j = 0; j < 3; ++j) {
    Wlin[j] = (const float*)d_in[21 + 4 * j];
    blin[j] = (const float*)d_in[22 + 4 * j];
    Wpp[j]  = (const float*)d_in[23 + 4 * j];
    bpp[j]  = (const float*)d_in[24 + 4 * j];
  }
  const float* Wfc1 = (const float*)d_in[33];
  const float* bfc1 = (const float*)d_in[34];
  const float* Wfc2 = (const float*)d_in[35];
  const float* bfc2 = (const float*)d_in[36];

  // deterministic bump allocator over d_ws
  char* base = (char*)d_ws; size_t off = 0;
  auto alloc = [&](size_t b) -> void* {
    void* p = base + off; off = (off + b + 255) & ~(size_t)255; return p;
  };
  float* X0 = (float*)alloc((size_t)NN * HF * 4);
  float* X1 = (float*)alloc((size_t)NN * HF * 4);
  float* H1 = (float*)alloc((size_t)NN * HF * 4);
  float* H2 = (float*)alloc((size_t)NN * HF * 4);
  float* HB = (float*)alloc((size_t)NN * HF * 4);
  float* S  = (float*)alloc((size_t)(NN + 1) * HF * 4);
  float* CNT = (float*)alloc((size_t)(NN + 1) * 4);
  int* SRC0 = (int*)alloc((size_t)EE * 4);
  int* DST0 = (int*)alloc((size_t)EE * 4);
  int* SRC1 = (int*)alloc((size_t)EE * 4);
  int* DST1 = (int*)alloc((size_t)EE * 4);
  int* B0 = (int*)alloc((size_t)NN * 4);
  int* B1 = (int*)alloc((size_t)NN * 4);
  float* RAW = (float*)alloc((size_t)EE * 4);
  float* ESC = (float*)alloc((size_t)EE * 4);
  float* SCO = (float*)alloc((size_t)EE * 4);
  unsigned* MU = (unsigned*)alloc((size_t)(NN + 1) * 4);
  float* Z = (float*)alloc((size_t)(NN + 1) * 4);
  unsigned* KEY = (unsigned*)alloc((size_t)E2 * 4);
  int* IDX = (int*)alloc((size_t)E2 * 4);
  int* CL = (int*)alloc((size_t)NN * 4);
  unsigned char* REM = (unsigned char*)alloc((size_t)NN);
  float* CS = (float*)alloc((size_t)NN * 4);
  float* GSUM = (float*)alloc((size_t)GG * HF * 4);
  float* GCNT = (float*)alloc((size_t)GG * 4);
  float* GV = (float*)alloc((size_t)GG * HF * 4);
  float* GH = (float*)alloc((size_t)GG * HF * 4);
  _Float16* PK = (_Float16*)alloc((size_t)19 * 16384 * 2);
  auto pk = [&](int i) { return PK + (size_t)i * 16384; };

  // pack all 128-wide weight slices into WMMA B fragments
  for (int j = 0; j < 6; ++j) {
    k_pack<<<64, 256, 0, stream>>>(Wl[j], 128, 0,   pk(2 * j));
    k_pack<<<64, 256, 0, stream>>>(Wr[j], 128, 0,   pk(2 * j + 1));
  }
  for (int j = 0; j < 3; ++j) {
    k_pack<<<64, 256, 0, stream>>>(Wlin[j], 256, 0,   pk(12 + 2 * j));   // cols 0:128 (h2)
    k_pack<<<64, 256, 0, stream>>>(Wlin[j], 256, 128, pk(13 + 2 * j));   // cols 128:256 (h1)
  }
  k_pack<<<64, 256, 0, stream>>>(Wfc1, 128, 0, pk(18));

  k_copy_f32<<<cdiv(NN * HF, 256), 256, 0, stream>>>(X0, Xin, NN * HF);
  k_copy_i32<<<cdiv(EE, 256), 256, 0, stream>>>(SRC0, EI, EE);
  k_copy_i32<<<cdiv(EE, 256), 256, 0, stream>>>(DST0, EI + EE, EE);
  k_copy_i32<<<cdiv(NN, 256), 256, 0, stream>>>(B0, Bin, NN);
  k_fill_f32<<<cdiv(GG * HF, 256), 256, 0, stream>>>(GSUM, 0.f, GG * HF);
  k_fill_f32<<<1, 256, 0, stream>>>(GCNT, 0.f, GG);

  float* cx = X0; float* nx = X1;
  int *cs = SRC0, *cd = DST0, *ns = SRC1, *nd = DST1;
  int *cb = B0, *nb = B1;

  auto runsort = [&]() {
    for (unsigned kk = 2; kk <= (unsigned)E2; kk <<= 1)
      for (unsigned jj = kk >> 1; jj > 0; jj >>= 1)
        k_bitonic<<<E2 / 256, 256, 0, stream>>>(KEY, IDX, (int)jj, (int)kk);
  };
  auto sage = [&](const float* xin, int conv, float* outp) {
    k_fill_f32<<<cdiv((NN + 1) * HF, 256), 256, 0, stream>>>(S, 0.f, (NN + 1) * HF);
    k_fill_f32<<<cdiv(NN + 1, 256), 256, 0, stream>>>(CNT, 0.f, NN + 1);
    k_edge_scatter<<<EE, 128, 0, stream>>>(cs, cd, xin, S, CNT);
    k_mean<<<cdiv(NN * HF, 256), 256, 0, stream>>>(S, CNT);
    k_gemm128<<<cdiv(NN, 16), 256, 0, stream>>>(
        S, pk(2 * conv), xin, pk(2 * conv + 1), bl[conv], outp, NN, 1);
  };

  for (int it = 0; it < 3; ++it) {
    sage(cx, 2 * it,     H1);                       // h1 = relu(sage(x))
    sage(H1, 2 * it + 1, H2);                       // h2 = relu(sage(h1))
    // h = concat([h2,h1]) @ Wlin.T + blin  (split into two K=128 passes)
    k_gemm128<<<cdiv(NN, 16), 256, 0, stream>>>(
        H2, pk(12 + 2 * it), H1, pk(13 + 2 * it), blin[it], HB, NN, 0);
    k_gaccum<<<cdiv(NN * HF, 256), 256, 0, stream>>>(HB, cb, GSUM, GCNT);

    // ---- edge pooling ----
    k_edge_score<<<EE / 8, 256, 0, stream>>>(HB, cs, cd, Wpp[it], bpp[it], RAW);
    k_fill_u32<<<cdiv(NN + 1, 256), 256, 0, stream>>>(MU, 0u, NN + 1);
    k_fill_f32<<<cdiv(NN + 1, 256), 256, 0, stream>>>(Z, 0.f, NN + 1);
    k_seg_max<<<cdiv(EE, 256), 256, 0, stream>>>(RAW, cd, MU);
    k_seg_exp<<<cdiv(EE, 256), 256, 0, stream>>>(RAW, cd, MU, Z, ESC);
    k_seg_norm<<<cdiv(EE, 256), 256, 0, stream>>>(ESC, cd, Z, SCO);
    k_build_skey<<<cdiv(E2, 256), 256, 0, stream>>>(SCO, cd, KEY, IDX);
    runsort();
    k_fill_u8<<<cdiv(NN, 256), 256, 0, stream>>>(REM, (unsigned char)1, NN);
    k_fill_f32<<<cdiv(NN, 256), 256, 0, stream>>>(CS, 1.f, NN);
    k_greedy<<<1, 1, 0, stream>>>(IDX, cs, cd, SCO, REM, CL, CS);
    k_fill_f32<<<cdiv(NN * HF, 256), 256, 0, stream>>>(nx, 0.f, NN * HF);
    k_cluster_scatter<<<cdiv(NN * HF, 256), 256, 0, stream>>>(HB, CL, nx);
    k_scale<<<cdiv(NN * HF, 256), 256, 0, stream>>>(nx, CS);
    k_fill_i32<<<cdiv(NN, 256), 256, 0, stream>>>(nb, GG, NN);
    k_batch_set<<<cdiv(NN, 256), 256, 0, stream>>>(CL, cb, nb);
    k_rekey<<<cdiv(E2, 256), 256, 0, stream>>>(cs, cd, CL, KEY, IDX);
    runsort();
    k_dedup<<<cdiv(EE, 256), 256, 0, stream>>>(KEY, ns, nd);

    { float* t = cx; cx = nx; nx = t; }
    { int* t = cs; cs = ns; ns = t; }
    { int* t = cd; cd = nd; nd = t; }
    { int* t = cb; cb = nb; nb = t; }
  }

  k_gaccum<<<cdiv(NN * HF, 256), 256, 0, stream>>>(cx, cb, GSUM, GCNT);
  k_gmean<<<cdiv(GG * HF, 256), 256, 0, stream>>>(GSUM, GCNT, GV);
  k_gemm128<<<cdiv(GG, 16), 256, 0, stream>>>(
      GV, pk(18), nullptr, nullptr, bfc1, GH, GG, 1);
  k_fc2<<<GG, 32, 0, stream>>>(GH, Wfc2, bfc2, (float*)d_out);
}
